// OptimizedGCN_59536836657839
// MI455X (gfx1250) — matrix operand order repaired and checked by reading, not compiled, not run on hardware
//
#include <hip/hip_runtime.h>
#include <stdint.h>

// ---------------------------------------------------------------------------
// GCN 3-layer inference for MI455X (gfx1250, wave32).
//   per layer:  H = Xbf16 @ Wbf16   (WMMA f32_16x16x32_bf16, f32 accum,
//                                    A staged in LDS, B frags register-resident)
//               AGG = D^-1/2 A D^-1/2 H  (self-loop init + per-edge f32 atomics,
//                                         output resident in 192MB L2)
//               act = relu(AGG + b) -> bf16 for next layer (fused pass)
// ---------------------------------------------------------------------------

#define F   128        // F_IN == F_HID == F_OUT == 128
#define TM  16         // WMMA tile M
#define KB  4          // 128 / 32 K-blocks per tile
#define MT  4          // M-tiles per block (64 rows staged in LDS)
#define LDP (F + 8)    // LDS row pitch in ushorts (+16B pad: conflict-free frags)

typedef __attribute__((ext_vector_type(16))) __bf16 v16bf;
typedef __attribute__((ext_vector_type(8)))  float  v8f;

union Frag16 { uint4 u[2]; v16bf v; };   // 32 bytes = 16 bf16 = 8 VGPRs

__device__ __forceinline__ unsigned short f32_to_bf16_rne(float f) {
    unsigned u = __float_as_uint(f);
    unsigned r = (u + 0x7FFFu + ((u >> 16) & 1u)) >> 16;   // round-to-nearest-even
    return (unsigned short)r;
}

// ----- small elementwise kernels -------------------------------------------

__global__ __launch_bounds__(256) void k_f32_to_bf16(const float* __restrict__ src,
                                                     unsigned short* __restrict__ dst,
                                                     int n) {
    int i = blockIdx.x * 256 + threadIdx.x;
    if (i < n) dst[i] = f32_to_bf16_rne(src[i]);
}

// Wt_bf16[n*128 + k] = bf16( W[k*128 + n] )  (transpose: B-fragment loads contiguous)
__global__ __launch_bounds__(256) void k_convWT(const float* __restrict__ W,
                                                unsigned short* __restrict__ WT) {
    int i = blockIdx.x * 256 + threadIdx.x;   // i = n*128 + k
    if (i < F * F) {
        int n = i >> 7, k = i & 127;
        WT[i] = f32_to_bf16_rne(W[k * F + n]);
    }
}

__global__ __launch_bounds__(256) void k_zero_f32(float* __restrict__ p, int n) {
    int i = blockIdx.x * 256 + threadIdx.x;
    if (i < n) p[i] = 0.0f;
}

__global__ __launch_bounds__(256) void k_deg_count(const int* __restrict__ col,
                                                   float* __restrict__ deg, int E) {
    int e = blockIdx.x * 256 + threadIdx.x;
    if (e < E) atomicAdd(&deg[col[e]], 1.0f);
}

__global__ __launch_bounds__(256) void k_dinv(const float* __restrict__ deg,
                                              float* __restrict__ dinv, int n) {
    int i = blockIdx.x * 256 + threadIdx.x;
    if (i < n) dinv[i] = rsqrtf(deg[i] + 1.0f);   // +1 self-loop; always > 0
}

// ----- WMMA GEMM: C(nrows,128) = A(nrows,128) * W(128,128) -----------------
// A: bf16 row-major, nrows % 16 == 0. BT: bf16, BT[n*128+k] = W[k][n].
// Block = 8 waves; block stages a 64-row A strip in LDS, wave w owns column
// tile w and keeps its 4 B fragments in registers across all 4 M-tiles.

__global__ __launch_bounds__(256) void k_gemm_wmma_bf16(
        const unsigned short* __restrict__ A,
        const unsigned short* __restrict__ BT,
        float* __restrict__ C,
        int nrows) {
    __shared__ alignas(16) unsigned short As[MT * TM * LDP];   // 17 KB

    const int nTiles = nrows >> 4;
    const int tile0  = blockIdx.x * MT;

    // cooperative global -> LDS: 64 rows x 128 bf16 = 1024 x 16B chunks
    {
        const uint4* src = reinterpret_cast<const uint4*>(A);  // 16B = 8 ushorts
        #pragma unroll
        for (int t = 0; t < 4; ++t) {
            int chunk = t * 256 + threadIdx.x;   // 0..1023
            int lrow  = chunk >> 4;              // local row 0..63
            int cpos  = chunk & 15;              // 16B unit within row
            int grow  = tile0 * TM + lrow;
            if (grow >= nrows) grow = nrows - 1; // clamp (tail blocks only)
            uint4* dst = reinterpret_cast<uint4*>(As + lrow * LDP) + cpos;
            *dst = src[(size_t)grow * (F / 8) + cpos];
        }
    }
    __syncthreads();

    const int wave = threadIdx.x >> 5;    // 0..7 -> column tile
    const int lane = threadIdx.x & 31;
    const int half = lane >> 4;           // 0: lanes 0-15, 1: lanes 16-31
    const int m16  = lane & 15;

    // B 16-bit 32x16 layout: lanes0-15 K {0..7,16..23}; lanes16-31 {8..15,24..31}
    // Load once, reuse for all MT tiles.
    Frag16 b[KB];
    const unsigned short* bcol = BT + (size_t)(wave * TM + m16) * F;
    #pragma unroll
    for (int kb = 0; kb < KB; ++kb) {
        b[kb].u[0] = *reinterpret_cast<const uint4*>(bcol + kb * 32 + half * 8);
        b[kb].u[1] = *reinterpret_cast<const uint4*>(bcol + kb * 32 + 16 + half * 8);
    }

    const int outcol = wave * TM + m16;
    #pragma unroll
    for (int t = 0; t < MT; ++t) {
        if (tile0 + t >= nTiles) break;   // uniform tail guard (rows % 16 == 0)

        // A 16-bit 16x32 layout: lanes0-15 K {0..7,16..23}; lanes16-31 {8..15,24..31}
        const unsigned short* arow = As + (t * TM + m16) * LDP;
        v8f c = {};
        #pragma unroll
        for (int kb = 0; kb < KB; ++kb) {
            Frag16 a;
            a.u[0] = *reinterpret_cast<const uint4*>(arow + kb * 32 + half * 8);
            a.u[1] = *reinterpret_cast<const uint4*>(arow + kb * 32 + 16 + half * 8);
            c = __builtin_amdgcn_wmma_f32_16x16x32_bf16(
                    /*neg_a=*/false, a.v, /*neg_b=*/false, b[kb].v,
                    /*c_mod=*/(short)0, c, /*reuse_a=*/false, /*reuse_b=*/false);
        }

        // C/D f32 layout: VGPR v -> M = v + half*8, N = m16; branchless stores
        float* crow = C + (size_t)((tile0 + t) * TM + half * 8) * F + outcol;
        #pragma unroll
        for (int v = 0; v < 8; ++v) crow[(size_t)v * F] = c[v];
    }
}

// ----- aggregation ----------------------------------------------------------

// Self-loop term initializes AGG fully (no pre-zero needed): AGG = H * dinv[i]^2
__global__ __launch_bounds__(256) void k_self_init(const float* __restrict__ H,
                                                   const float* __restrict__ dinv,
                                                   float* __restrict__ AGG, int nf) {
    int i = blockIdx.x * 256 + threadIdx.x;
    if (i < nf) {
        float d = dinv[i >> 7];
        AGG[i] = H[i] * d * d;
    }
}

// One wave per edge; each lane handles 4 features (float4). Output buffer is
// 51.2MB < 192MB L2, so the f32 atomic RMWs stay in L2.
__global__ __launch_bounds__(256) void k_edge_agg(const float* __restrict__ H,
                                                  const int* __restrict__ row,
                                                  const int* __restrict__ col,
                                                  const float* __restrict__ dinv,
                                                  float* __restrict__ AGG, int E) {
    int e = blockIdx.x * 8 + (threadIdx.x >> 5);
    if (e >= E) return;
    int lane = threadIdx.x & 31;
    int r = row[e], c = col[e];
    float ew = dinv[r] * dinv[c];
    const float4 v = *reinterpret_cast<const float4*>(H + (size_t)r * F + lane * 4);
    float* o = AGG + (size_t)c * F + lane * 4;
    atomicAdd(o + 0, v.x * ew);
    atomicAdd(o + 1, v.y * ew);
    atomicAdd(o + 2, v.z * ew);
    atomicAdd(o + 3, v.w * ew);
}

// Hidden layers: act = relu(AGG + b) written as bf16 for the next GEMM.
__global__ __launch_bounds__(256) void k_post_relu_bf16(const float* __restrict__ AGG,
                                                        const float* __restrict__ bias,
                                                        unsigned short* __restrict__ HBF,
                                                        int nf) {
    int i = blockIdx.x * 256 + threadIdx.x;
    if (i < nf) {
        float v = AGG[i] + bias[i & 127];
        v = v > 0.0f ? v : 0.0f;
        HBF[i] = f32_to_bf16_rne(v);
    }
}

// Final layer: in-place bias add on d_out (no activation).
__global__ __launch_bounds__(256) void k_post_bias(float* __restrict__ OUT,
                                                   const float* __restrict__ bias,
                                                   int nf) {
    int i = blockIdx.x * 256 + threadIdx.x;
    if (i < nf) OUT[i] += bias[i & 127];
}

// ---------------------------------------------------------------------------

extern "C" void kernel_launch(void* const* d_in, const int* in_sizes, int n_in,
                              void* d_out, int out_size, void* d_ws, size_t ws_size,
                              hipStream_t stream) {
    const float* x  = (const float*)d_in[0];
    const float* W1 = (const float*)d_in[1];
    const float* b1 = (const float*)d_in[2];
    const float* W2 = (const float*)d_in[3];
    const float* b2 = (const float*)d_in[4];
    const float* W3 = (const float*)d_in[5];
    const float* b3 = (const float*)d_in[6];
    const int*   ei = (const int*)d_in[7];

    const int Nn = in_sizes[0] / F;     // 100000
    const int E  = in_sizes[7] / 2;     // 1600000
    const int nf = Nn * F;              // 12.8M elements
    const int* rowp = ei;
    const int* colp = ei + E;

    // workspace layout (256B-aligned offsets)
    char* ws = (char*)d_ws;
    const size_t SZ_HBF = (size_t)nf * 2;                 // 25.6 MB bf16 activations
    const size_t SZ_F32 = (size_t)nf * 4;                 // 51.2 MB
    size_t off = 0;
    unsigned short* HBF = (unsigned short*)(ws + off); off += (SZ_HBF + 255) & ~255ull;
    float* Hf  = (float*)(ws + off);                   off += (SZ_F32 + 255) & ~255ull;
    float* AGG = (float*)(ws + off);                   off += (SZ_F32 + 255) & ~255ull;
    float* DEG = (float*)(ws + off);                   off += ((size_t)Nn * 4 + 255) & ~255ull;
    float* DNV = (float*)(ws + off);                   off += ((size_t)Nn * 4 + 255) & ~255ull;
    unsigned short* WBF = (unsigned short*)(ws + off); // 32 KB

    const int B = 256;
    const int gNF = (nf + B - 1) / B;
    const int gN  = (Nn + B - 1) / B;
    const int gE  = (E + B - 1) / B;
    const int gW  = (F * F + B - 1) / B;
    const int nTiles = Nn / TM;                 // rows % 16 == 0 (100000/16 = 6250)
    const int gM  = (nTiles + MT - 1) / MT;     // 1563 blocks, 4 tiles each
    const int gEd = (E + 7) / 8;                // 8 edges per 256-thread block

    // normalization: deg = in-degree + 1 (self loop), dinv = rsqrt(deg)
    k_zero_f32 <<<gN,  B, 0, stream>>>(DEG, Nn);
    k_deg_count<<<gE,  B, 0, stream>>>(colp, DEG, E);
    k_dinv     <<<gN,  B, 0, stream>>>(DEG, DNV, Nn);

    // bf16 copy of input features
    k_f32_to_bf16<<<gNF, B, 0, stream>>>(x, HBF, nf);

    auto layer = [&](const float* W, const float* bias, bool relu, float* aggOut) {
        k_convWT        <<<gW,  B, 0, stream>>>(W, WBF);
        k_gemm_wmma_bf16<<<gM,  B, 0, stream>>>(HBF, WBF, Hf, Nn);
        k_self_init     <<<gNF, B, 0, stream>>>(Hf, DNV, aggOut, nf);
        k_edge_agg      <<<gEd, B, 0, stream>>>(Hf, rowp, colp, DNV, aggOut, E);
        if (relu) k_post_relu_bf16<<<gNF, B, 0, stream>>>(aggOut, bias, HBF, nf);
        else      k_post_bias     <<<gNF, B, 0, stream>>>(aggOut, bias, nf);
    };

    layer(W1, b1, /*relu=*/true,  AGG);
    layer(W2, b2, /*relu=*/true,  AGG);
    layer(W3, b3, /*relu=*/false, (float*)d_out);
}
